// MultiHeadAttention_1700807049608
// MI455X (gfx1250) — matrix operand order repaired and checked by reading, not compiled, not run on hardware
//
#include <hip/hip_runtime.h>
#include <hip/hip_bf16.h>

#define B_ 4
#define S_ 2048
#define F_ 1024
#define H_ 16
#define D_ 64

typedef __bf16 bf16;
typedef __attribute__((ext_vector_type(16))) __bf16 v16bf;
typedef __attribute__((ext_vector_type(8)))  __bf16 v8bf;
typedef __attribute__((ext_vector_type(8)))  float  v8f;
typedef int v4i_vec __attribute__((__vector_size__(16)));

// ---------------------------------------------------------------------------
// Fragment loaders (CDNA5 16-bit WMMA layouts, CDNA5 ISA §7.12.2)
// A 16x32: lane holds row M=lane&15; K chunks {kb..kb+7} and {kb+16..kb+23},
//          kb = 8*(lane>=16).
// B 32x16: lane holds col N=lane&15; K chunks {kb..kb+7} and {kb+16..kb+23},
//          kb = 8*(lane>=16).
// ---------------------------------------------------------------------------
__device__ __forceinline__ v16bf frag_ld_a(const bf16* p, int lane) {
  const int kb = (lane >> 4) << 3;
  union { uint4 u; v8bf v; } c0, c1;
  c0.u = *(const uint4*)(p + kb);
  c1.u = *(const uint4*)(p + kb + 16);
  v16bf r;
#pragma unroll
  for (int i = 0; i < 8; ++i) { r[i] = c0.v[i]; r[i + 8] = c1.v[i]; }
  return r;
}

__device__ __forceinline__ v16bf frag_ld_b(const bf16* p, int lane) {
  const int kb = (lane >> 4) << 3;
  union { uint4 u; v8bf v; } c0, c1;
  c0.u = *(const uint4*)(p + kb);
  c1.u = *(const uint4*)(p + kb + 16);
  v16bf r;
#pragma unroll
  for (int i = 0; i < 8; ++i) { r[i] = c0.v[i]; r[i + 8] = c1.v[i]; }
  return r;
}

// A-position fragment built from f32 LDS data (softmax probabilities)
__device__ __forceinline__ v16bf frag_cvt_a_f32(const float* p, int lane) {
  const int kb = (lane >> 4) << 3;
  float4 f0 = *(const float4*)(p + kb);
  float4 f1 = *(const float4*)(p + kb + 4);
  float4 f2 = *(const float4*)(p + kb + 16);
  float4 f3 = *(const float4*)(p + kb + 20);
  v16bf r;
  r[0] = (bf16)f0.x; r[1] = (bf16)f0.y; r[2]  = (bf16)f0.z; r[3]  = (bf16)f0.w;
  r[4] = (bf16)f1.x; r[5] = (bf16)f1.y; r[6]  = (bf16)f1.z; r[7]  = (bf16)f1.w;
  r[8] = (bf16)f2.x; r[9] = (bf16)f2.y; r[10] = (bf16)f2.z; r[11] = (bf16)f2.w;
  r[12] = (bf16)f3.x; r[13] = (bf16)f3.y; r[14] = (bf16)f3.z; r[15] = (bf16)f3.w;
  return r;
}

__device__ __forceinline__ v8f wmma_bf16(v16bf a, v16bf b, v8f c) {
  return __builtin_amdgcn_wmma_f32_16x16x32_bf16(false, a, false, b, (short)0, c,
                                                 false, false);
}

// ---------------------------------------------------------------------------
// Async global->LDS staging (gfx1250 GLOBAL_LOAD_ASYNC_TO_LDS_B128, ASYNCcnt)
// with synchronous fallback if the builtin is unavailable.
// Probe-confirmed signature: param0 = AS1 pointer to int __vector_size__(16),
// param1 = AS3 pointer, then imm offset, imm cpol.
// ---------------------------------------------------------------------------
#if defined(__AMDGCN__) && __has_builtin(__builtin_amdgcn_global_load_async_to_lds_b128)
#define ASYNC_STAGE 1
#else
#define ASYNC_STAGE 0
#endif

__device__ __forceinline__ void stage_tile(const bf16* src, bf16* dst) {
#if ASYNC_STAGE
  __builtin_amdgcn_global_load_async_to_lds_b128(
      (__attribute__((address_space(1))) v4i_vec*)(
          (__attribute__((address_space(1))) void*)(src)),
      (__attribute__((address_space(3))) v4i_vec*)(
          (__attribute__((address_space(3))) void*)(dst)),
      0, 0);
#else
  *(uint4*)dst = *(const uint4*)src;
#endif
}

// ---------------------------------------------------------------------------
// Elementwise f32 -> bf16 conversion
// ---------------------------------------------------------------------------
__global__ void f32_to_bf16(const float* __restrict__ src, bf16* __restrict__ dst,
                            int n) {
  int i = (blockIdx.x * blockDim.x + threadIdx.x) * 4;
  if (i + 3 < n) {
    float4 f = *(const float4*)(src + i);
    dst[i]     = (bf16)f.x;
    dst[i + 1] = (bf16)f.y;
    dst[i + 2] = (bf16)f.z;
    dst[i + 3] = (bf16)f.w;
  } else {
    for (; i < n; ++i) dst[i] = (bf16)src[i];
  }
}

// ---------------------------------------------------------------------------
// GEMM: C[M,N] = A[M,K] @ W^T + bias   (W is [N,K] row-major, bf16 inputs)
// Block = 256 threads = 8 waves; block tile 256(M) x 64(N); wave tile 32x64.
// W tile [64][32] double-buffered in LDS via async-to-LDS staging.
// MODE 0: qh layout [B,H,S,D], scaled by 1/sqrt(D), bf16 out
// MODE 1: kh layout [B,H,S,D], bf16 out
// MODE 2: v transposed layout [B,H,D,S], bf16 out
// MODE 3: plain [M,N] f32 out with ReLU
// ---------------------------------------------------------------------------
template <int MODE>
__global__ void __launch_bounds__(256)
gemm_wmma(const bf16* __restrict__ A, const bf16* __restrict__ W,
          const float* __restrict__ bias, void* __restrict__ out) {
  __shared__ bf16 wtile[2][64][32];   // 8 KB, double buffered
  const int lane = threadIdx.x & 31;
  const int wave = threadIdx.x >> 5;
  const int m0 = blockIdx.y * 256 + wave * 32;
  const int n0 = blockIdx.x * 64;
  const int K = F_;

  // staging assignment: 256 threads x 16B == 64x32 bf16 tile
  const int srow = threadIdx.x >> 2;
  const int scol = (threadIdx.x & 3) * 8;
  const bf16* ssrc = W + (size_t)(n0 + srow) * K + scol;

  v8f acc[2][4] = {};
  const bf16* arow0 = A + (size_t)(m0 + (lane & 15)) * K;
  const bf16* arow1 = arow0 + (size_t)16 * K;

  int buf = 0;
  stage_tile(ssrc, &wtile[0][srow][scol]);
  for (int k0 = 0; k0 < K; k0 += 32) {
    if (k0 + 32 < K) {
      stage_tile(ssrc + k0 + 32, &wtile[buf ^ 1][srow][scol]);
#if ASYNC_STAGE
      __builtin_amdgcn_s_wait_asynccnt(1);   // current buffer's loads done
#endif
    } else {
#if ASYNC_STAGE
      __builtin_amdgcn_s_wait_asynccnt(0);
#endif
    }
    __syncthreads();                         // all threads' staging visible

    v16bf a0 = frag_ld_a(arow0 + k0, lane);
    v16bf a1 = frag_ld_a(arow1 + k0, lane);
#pragma unroll
    for (int j = 0; j < 4; ++j) {
      v16bf b = frag_ld_b(&wtile[buf][16 * j + (lane & 15)][0], lane);
      acc[0][j] = wmma_bf16(a0, b, acc[0][j]);
      acc[1][j] = wmma_bf16(a1, b, acc[1][j]);
    }
    __syncthreads();                         // done reading buf before restage
    buf ^= 1;
  }

#pragma unroll
  for (int half = 0; half < 2; ++half) {
    const int mlo = m0 + half * 16 + ((lane >> 4) << 3);
#pragma unroll
    for (int j = 0; j < 4; ++j) {
      const int n = n0 + 16 * j + (lane & 15);
      const float bn = bias[n];
#pragma unroll
      for (int r = 0; r < 8; ++r) {
        const int m = mlo + r;
        float v = acc[half][j][r] + bn;
        if constexpr (MODE == 3) {
          ((float*)out)[(size_t)m * F_ + n] = fmaxf(v, 0.0f);
        } else {
          const int b = m >> 11;            // m / S_
          const int s = m & (S_ - 1);
          const int h = n >> 6;             // n / D_
          const int d = n & (D_ - 1);
          if constexpr (MODE == 0) v *= 0.125f;  // 1/sqrt(64)
          size_t idx;
          if constexpr (MODE == 2)
            idx = ((size_t)(b * H_ + h) * D_ + d) * S_ + s;
          else
            idx = ((size_t)(b * H_ + h) * S_ + s) * D_ + d;
          ((bf16*)out)[idx] = (bf16)v;
        }
      }
    }
  }
}

// ---------------------------------------------------------------------------
// Attention: one block per (b, h, 16-row q tile).
// Phase 1: scores[16][2048] = qh_tile @ kh^T into LDS (WMMA)
// Phase 2: exact softmax per row (float4 passes); write attn (f32) to global
// Phase 3: ctx[16][64] = P @ V via WMMA (V stored [B,H,D,S]); bf16 concat out
// ---------------------------------------------------------------------------
__global__ void __launch_bounds__(256)
attn_kernel(const bf16* __restrict__ qh, const bf16* __restrict__ kh,
            const bf16* __restrict__ vhT, float* __restrict__ attn,
            bf16* __restrict__ ctx) {
  __shared__ float sc[16][S_];       // 128 KB
  __shared__ float red[8][16][16];   // 8 KB

  const int lane = threadIdx.x & 31;
  const int wave = threadIdx.x >> 5;
  const int bh = blockIdx.x >> 7;    // S_/16 == 128 q tiles per (b,h)
  const int qt = blockIdx.x & 127;
  const int q0 = qt * 16;

  const bf16* qb = qh + ((size_t)bh * S_ + q0) * D_;
  const bf16* kb = kh + (size_t)bh * S_ * D_;
  const bf16* vb = vhT + (size_t)bh * D_ * S_;
  float* attnb = attn + ((size_t)bh * S_ + q0) * S_;

  // ---- Phase 1: scores ----
  const bf16* qrow = qb + (size_t)(lane & 15) * D_;
  v16bf a0 = frag_ld_a(qrow, lane);
  v16bf a1 = frag_ld_a(qrow + 32, lane);
  for (int t = 0; t < 16; ++t) {
    const int kn0 = (wave + t * 8) * 16;
    const bf16* krow = kb + (size_t)(kn0 + (lane & 15)) * D_;
    v8f acc = {};
    acc = wmma_bf16(a0, frag_ld_b(krow, lane), acc);
    acc = wmma_bf16(a1, frag_ld_b(krow + 32, lane), acc);
    const int rlo = (lane >> 4) << 3;
#pragma unroll
    for (int r = 0; r < 8; ++r) sc[rlo + r][kn0 + (lane & 15)] = acc[r];
  }
  __syncthreads();

  // ---- Phase 2: softmax (2 rows per wave), float4 passes, write attn ----
  for (int row = wave * 2; row < wave * 2 + 2; ++row) {
    float mx = -3.4e38f;
    for (int c = lane * 4; c < S_; c += 128) {
      float4 f = *(const float4*)&sc[row][c];
      mx = fmaxf(mx, fmaxf(fmaxf(f.x, f.y), fmaxf(f.z, f.w)));
    }
#pragma unroll
    for (int o = 16; o > 0; o >>= 1) mx = fmaxf(mx, __shfl_xor(mx, o, 32));
    float sum = 0.0f;
    for (int c = lane * 4; c < S_; c += 128) {
      float4 f = *(const float4*)&sc[row][c];
      f.x = __expf(f.x - mx); f.y = __expf(f.y - mx);
      f.z = __expf(f.z - mx); f.w = __expf(f.w - mx);
      *(float4*)&sc[row][c] = f;
      sum += (f.x + f.y) + (f.z + f.w);
    }
#pragma unroll
    for (int o = 16; o > 0; o >>= 1) sum += __shfl_xor(sum, o, 32);
    const float inv = 1.0f / sum;
    float* arow = attnb + (size_t)row * S_;
    for (int c = lane * 4; c < S_; c += 128) {
      float4 f = *(const float4*)&sc[row][c];
      f.x *= inv; f.y *= inv; f.z *= inv; f.w *= inv;
      *(float4*)&sc[row][c] = f;
      *(float4*)(arow + c) = f;        // b128 store: attn is the BW floor
    }
  }
  __syncthreads();

  // ---- Phase 3: ctx = P @ V; wave -> (d-tile = wave&3, key half = wave>>2) ----
  const int d0 = (wave & 3) * 16;
  const int kbase = (wave >> 2) * (S_ / 2);
  v8f acc = {};
  const float* prow = &sc[lane & 15][0];
  const bf16* vrow = vb + (size_t)(d0 + (lane & 15)) * S_;
  for (int kt = kbase; kt < kbase + S_ / 2; kt += 32) {
    v16bf a = frag_cvt_a_f32(prow + kt, lane);
    v16bf b = frag_ld_b(vrow + kt, lane);
    acc = wmma_bf16(a, b, acc);
  }
  {
    const int rlo = (lane >> 4) << 3;
#pragma unroll
    for (int r = 0; r < 8; ++r) red[wave][rlo + r][lane & 15] = acc[r];
  }
  __syncthreads();

  // reduce the two key-halves; store ctx in concat layout [B,S,F]
  const int b = bh >> 4, h = bh & 15;
  for (int e = threadIdx.x; e < 16 * 64; e += 256) {
    const int m = e >> 6;
    const int n = e & 63;
    const int nt = n >> 4, nn = n & 15;
    float v = red[nt][m][nn] + red[nt + 4][m][nn];
    ctx[((size_t)b * S_ + q0 + m) * F_ + h * D_ + n] = (bf16)v;
  }
}

// ---------------------------------------------------------------------------
extern "C" void kernel_launch(void* const* d_in, const int* in_sizes, int n_in,
                              void* d_out, int out_size, void* d_ws, size_t ws_size,
                              hipStream_t stream) {
  (void)in_sizes; (void)n_in; (void)out_size; (void)ws_size;

  const float* q    = (const float*)d_in[0];
  const float* k    = (const float*)d_in[1];
  const float* v    = (const float*)d_in[2];
  const float* wq_w = (const float*)d_in[3];
  const float* wq_b = (const float*)d_in[4];
  const float* wk_w = (const float*)d_in[5];
  const float* wk_b = (const float*)d_in[6];
  const float* wv_w = (const float*)d_in[7];
  const float* wv_b = (const float*)d_in[8];
  const float* fc_w = (const float*)d_in[9];
  const float* fc_b = (const float*)d_in[10];

  const size_t MSZ = (size_t)B_ * S_ * F_;  // 8,388,608 elements
  const size_t WSZ = (size_t)F_ * F_;       // 1,048,576 elements

  bf16* p = (bf16*)d_ws;
  bf16* qbf  = p; p += MSZ;
  bf16* kbf  = p; p += MSZ;
  bf16* vbf  = p; p += MSZ;
  bf16* wqbf = p; p += WSZ;
  bf16* wkbf = p; p += WSZ;
  bf16* wvbf = p; p += WSZ;
  bf16* fcbf = p; p += WSZ;
  bf16* qhb  = p; p += MSZ;  // [B,H,S,D]
  bf16* khb  = p; p += MSZ;  // [B,H,S,D]
  bf16* vtb  = p; p += MSZ;  // [B,H,D,S]
  bf16* ctx  = p; p += MSZ;  // [B,S,F]

  float* outp = (float*)d_out;
  float* attn = outp + MSZ;

  auto cvt = [&](const float* s, bf16* d, size_t n) {
    f32_to_bf16<<<dim3((unsigned)((n / 4 + 255) / 256)), dim3(256), 0, stream>>>(
        s, d, (int)n);
  };
  cvt(q, qbf, MSZ);
  cvt(k, kbf, MSZ);
  cvt(v, vbf, MSZ);
  cvt(wq_w, wqbf, WSZ);
  cvt(wk_w, wkbf, WSZ);
  cvt(wv_w, wvbf, WSZ);
  cvt(fc_w, fcbf, WSZ);

  dim3 gblk(F_ / 64, (B_ * S_) / 256);
  gemm_wmma<0><<<gblk, 256, 0, stream>>>(qbf, wqbf, wq_b, qhb);
  gemm_wmma<1><<<gblk, 256, 0, stream>>>(kbf, wkbf, wk_b, khb);
  gemm_wmma<2><<<gblk, 256, 0, stream>>>(vbf, wvbf, wv_b, vtb);

  attn_kernel<<<B_ * H_ * (S_ / 16), 256, 0, stream>>>(qhb, khb, vtb, attn, ctx);

  gemm_wmma<3><<<gblk, 256, 0, stream>>>(ctx, fcbf, fc_b, d_out);
}
